// LSTM_5171140625003
// MI455X (gfx1250) — matrix operand order repaired and checked by reading, not compiled, not run on hardware
//
#include <hip/hip_runtime.h>

// Problem constants (match reference)
#define B_   64
#define S_   512
#define DIN_ 1024
#define HID_ 4096
#define HS_  1024

typedef __bf16 bhalf;
typedef bhalf  v16bf  __attribute__((ext_vector_type(16)));
typedef bhalf  bf16x8 __attribute__((ext_vector_type(8)));
typedef float  v8f    __attribute__((ext_vector_type(8)));
typedef int    v4i_   __attribute__((ext_vector_type(4)));

// ---------------- workspace layout (bytes) ----------------
#define OFF_XB    ((size_t)0)                         // x   bf16: 64 MB
#define OFF_WXB   (OFF_XB  + (size_t)B_*S_*DIN_*2)    // wx  bf16:  8 MB
#define OFF_WHB   (OFF_WXB + (size_t)HID_*DIN_*2)     // wh  bf16:  8 MB
#define OFF_BIAS  (OFF_WHB + (size_t)HID_*HS_*2)      // bias f32: 16 KB
#define OFF_XG    (OFF_BIAS + (size_t)HID_*4)         // xg  f32: 512 MB
#define OFF_H     (OFF_XG  + (size_t)S_*B_*HID_*4)    // h   bf16 double buffer
#define OFF_C     (OFF_H   + (size_t)2*B_*HS_*2)      // c   f32

// ---------------- CDNA5 async global->LDS path ----------------
#if defined(__HIP_DEVICE_COMPILE__) && defined(__gfx1250__) && \
    __has_builtin(__builtin_amdgcn_global_load_async_to_lds_b128)
#define ASYNC_LDS 1
#else
#define ASYNC_LDS 0
#endif

// Copy 16 B from global to LDS. Async variant is tracked by ASYNCcnt and
// runs concurrently with compute; fallback is a plain load/ds_store pair.
// The builtin takes generic pointers to 128-bit vectors (per hipcc's own
// diagnostic of its prototype).
__device__ __forceinline__ void copy16_g2l(const void* gsrc, void* ldst) {
#if ASYNC_LDS
    __builtin_amdgcn_global_load_async_to_lds_b128(
        (v4i_*)gsrc, (v4i_*)ldst, 0, 0);
#else
    *(uint4*)ldst = *(const uint4*)gsrc;
#endif
}

__device__ __forceinline__ void async_stage_sync() {
#if ASYNC_LDS
#if __has_builtin(__builtin_amdgcn_s_wait_asynccnt)
    __builtin_amdgcn_s_wait_asynccnt(0);
#else
    asm volatile("s_wait_asynccnt 0" ::: "memory");
#endif
#endif
    __syncthreads();
}

// Load 16 contiguous bf16 (32 B) into a B-matrix fragment (works for
// global or LDS pointers; compiler picks global_load_b128 / ds_load).
__device__ __forceinline__ v16bf load_frag16(const bhalf* p) {
    bf16x8 lo = *(const bf16x8*)(p);
    bf16x8 hi = *(const bf16x8*)(p + 8);
    return __builtin_shufflevector(lo, hi,
        0,1,2,3,4,5,6,7,8,9,10,11,12,13,14,15);
}

// A-matrix 16x32 bf16 fragment: lane m = lane&15, kh = lane>>4.
// VGPR0..3 hold K = kh*16 + 0..7 ; VGPR4..7 hold K = kh*16 + 8..15.
__device__ __forceinline__ v16bf load_frag_a(const bhalf* __restrict__ row,
                                             int kb, int kh) {
    bf16x8 lo = *(const bf16x8*)(row + kb + kh * 16);
    bf16x8 hi = *(const bf16x8*)(row + kb + kh * 16 + 8);
    return __builtin_shufflevector(lo, hi,
        0,1,2,3,4,5,6,7,8,9,10,11,12,13,14,15);
}

__device__ __forceinline__ float sigmoid_(float x) {
    return 1.0f / (1.0f + __expf(-x));
}
__device__ __forceinline__ float tanh_(float x) {
    float xx = fminf(fmaxf(x, -15.0f), 15.0f);
    float e  = __expf(2.0f * xx);
    return (e - 1.0f) / (e + 1.0f);
}

// ---------------- kernels ----------------

__global__ void f32_to_bf16_kernel(const float* __restrict__ in,
                                   bhalf* __restrict__ out, int n) {
    int i = blockIdx.x * blockDim.x + threadIdx.x;
    int stride = gridDim.x * blockDim.x;
    for (; i < n; i += stride) out[i] = (bhalf)in[i];
}

// bias = bx + bh ; zero h0 (bf16 buffer 0) and c0
__global__ void prep_kernel(const float* __restrict__ bx,
                            const float* __restrict__ bh,
                            float* __restrict__ bias,
                            bhalf* __restrict__ h0,
                            float* __restrict__ c0) {
    int i = blockIdx.x * blockDim.x + threadIdx.x;
    if (i < HID_) bias[i] = bx[i] + bh[i];
    if (i < B_ * HS_) { h0[i] = (bhalf)0.0f; c0[i] = 0.0f; }
}

// xg[s][b][g] = sum_d x[b][s][d] * wx[g][d] + bias[g]
// One wave -> 32(M) x 64(N) tile: 8 accumulators, every B fragment
// feeds two WMMAs (M reuse) -> ~2x arithmetic intensity vs 16x64.
__global__ __launch_bounds__(128)
void xg_gemm_kernel(const bhalf* __restrict__ xb,
                    const bhalf* __restrict__ wxb,
                    const float* __restrict__ bias,
                    float* __restrict__ xg) {
    const int lane = threadIdx.x & 31;
    const int wave = blockIdx.x * 4 + (threadIdx.x >> 5);
    const int nT   = HID_ / 64;            // 64
    const int mt   = wave / nT;            // 0..1023 (32-row tiles)
    const int nt   = wave % nT;
    const int l15  = lane & 15;
    const int kh   = lane >> 4;

    const bhalf* arow0 = xb + (size_t)(mt * 32 + l15) * DIN_;
    const bhalf* arow1 = xb + (size_t)(mt * 32 + 16 + l15) * DIN_;
    const bhalf* brow[4];
#pragma unroll
    for (int j = 0; j < 4; ++j)
        brow[j] = wxb + (size_t)(nt * 64 + j * 16 + l15) * DIN_;

    v8f acc[2][4] = {};
    for (int kb = 0; kb < DIN_; kb += 32) {
        v16bf a0 = load_frag_a(arow0, kb, kh);
        v16bf a1 = load_frag_a(arow1, kb, kh);
#pragma unroll
        for (int j = 0; j < 4; ++j) {
            v16bf b = load_frag16(brow[j] + kb + kh * 16);
            acc[0][j] = __builtin_amdgcn_wmma_f32_16x16x32_bf16(
                false, a0, false, b, (short)0, acc[0][j], false, false);
            acc[1][j] = __builtin_amdgcn_wmma_f32_16x16x32_bf16(
                false, a1, false, b, (short)0, acc[1][j], false, false);
        }
    }

#pragma unroll
    for (int j = 0; j < 4; ++j) {
        const int n  = nt * 64 + j * 16 + l15;
        const float bn = bias[n];
#pragma unroll
        for (int h2 = 0; h2 < 2; ++h2) {
#pragma unroll
            for (int i = 0; i < 8; ++i) {
                const int r  = mt * 32 + h2 * 16 + i + 8 * kh; // flattened (b,s)
                const int bb = r >> 9;                         // r / S_
                const int ss = r & (S_ - 1);                   // r % S_
                xg[((size_t)ss * B_ + bb) * HID_ + n] = acc[h2][j][i] + bn;
            }
        }
    }
}

// One LSTM step, fully fused: gates = xg_t + h_in * wh^T, then elementwise.
// Block = 8 waves covering all 4 batch tiles x 2 hidden tiles. The wh
// B-tile (128 rows x K) is staged through LDS: 4 stages of 64 KB, double
// buffered (128 KB), with CDNA5 async global->LDS copies prefetching the
// next stage while WMMAs consume the current one.
__global__ __launch_bounds__(256)
void lstm_step_kernel(const float* __restrict__ xg,
                      const bhalf* __restrict__ whb,
                      const bhalf* __restrict__ h_in,
                      bhalf* __restrict__ h_out,
                      float* __restrict__ c,
                      float* __restrict__ out,
                      int s) {
    extern __shared__ char smem[];
    bhalf* lB = (bhalf*)smem;           // 2 x (128 rows x 256 K) bf16 = 128 KB
    const int BUFELE = 128 * 256;       // elements per buffer

    const int tid  = threadIdx.x;
    const int lane = tid & 31;
    const int w    = tid >> 5;          // local wave 0..7
    const int mt   = w & 3;             // batch tile
    const int jtl  = w >> 2;            // local hidden tile 0..1
    const int jt   = blockIdx.x * 2 + jtl;   // 0..63
    const int l15  = lane & 15;
    const int kh   = lane >> 4;

    const bhalf* arow = h_in + (size_t)(mt * 16 + l15) * HS_;

    // Cooperative staging of one 64 KB stage (K range [ks, ks+128)).
    // local row lr = ljt*64 + q*16 + cl  maps to wh row q*HS + jt*16 + cl.
    auto stage_copy = [&](bhalf* buf, int ks) {
        for (int cid = tid; cid < 4096; cid += 256) {
            const int lr  = cid >> 4;        // 0..255
            const int c16 = cid & 15;        // 16B chunk within row segment
            const int ljt = lr >> 6;
            const int q   = (lr >> 4) & 3;
            const int cl  = lr & 15;
            const int g   = q * HS_ + (blockIdx.x * 2 + ljt) * 16 + cl;
            const uint4* src = (const uint4*)(whb + (size_t)g * HS_ + ks) + c16;
            uint4* dst = (uint4*)(buf + lr * 256) + c16;
            copy16_g2l(src, dst);
        }
    };

    v8f acc[4] = {};
    int cur = 0;
    stage_copy(lB, 0);
    async_stage_sync();

    for (int st = 0; st < 4; ++st) {
        if (st + 1 < 4)
            stage_copy(lB + (1 - cur) * BUFELE, (st + 1) * 256);

        const bhalf* lbase = lB + cur * BUFELE + (size_t)jtl * 64 * 256;
        const int ks = st * 256;
        for (int kb = 0; kb < 256; kb += 32) {
            v16bf a = load_frag_a(arow, ks + kb, kh);
#pragma unroll
            for (int q = 0; q < 4; ++q) {
                v16bf b = load_frag16(lbase + ((size_t)q * 16 + l15) * 256
                                      + kb + kh * 16);
                acc[q] = __builtin_amdgcn_wmma_f32_16x16x32_bf16(
                    false, a, false, b, (short)0, acc[q], false, false);
            }
        }

        if (st + 1 < 4) async_stage_sync();
        cur ^= 1;
    }

    const int j = jt * 16 + l15;
#pragma unroll
    for (int i = 0; i < 8; ++i) {
        const int b = mt * 16 + i + 8 * kh;
        const size_t xbase = ((size_t)s * B_ + b) * HID_;
        const float gi = acc[0][i] + xg[xbase + 0 * HS_ + j];
        const float gf = acc[1][i] + xg[xbase + 1 * HS_ + j];
        const float gg = acc[2][i] + xg[xbase + 2 * HS_ + j];
        const float go = acc[3][i] + xg[xbase + 3 * HS_ + j];

        const float iv = sigmoid_(gi);
        const float fv = sigmoid_(gf);
        const float gv = tanh_(gg);
        const float ov = sigmoid_(go);

        const size_t cidx = (size_t)b * HS_ + j;
        const float cn = fv * c[cidx] + iv * gv;
        c[cidx] = cn;
        const float hv = ov * tanh_(cn);

        out[((size_t)b * S_ + s) * HS_ + j] = hv;   // out[b][s][j]
        h_out[cidx] = (bhalf)hv;

        if (s == S_ - 1) {
            const size_t mainsz = (size_t)B_ * S_ * HS_;
            out[mainsz + cidx] = hv;                        // final h
            out[mainsz + (size_t)B_ * HS_ + cidx] = cn;     // final c
        }
    }
}

// ---------------- host launcher ----------------
extern "C" void kernel_launch(void* const* d_in, const int* in_sizes, int n_in,
                              void* d_out, int out_size, void* d_ws, size_t ws_size,
                              hipStream_t stream) {
    const float* x  = (const float*)d_in[0];
    const float* wx = (const float*)d_in[1];
    const float* wh = (const float*)d_in[2];
    const float* bx = (const float*)d_in[3];
    const float* bh = (const float*)d_in[4];
    float* out = (float*)d_out;
    char*  ws  = (char*)d_ws;

    bhalf* xb   = (bhalf*)(ws + OFF_XB);
    bhalf* wxb  = (bhalf*)(ws + OFF_WXB);
    bhalf* whb  = (bhalf*)(ws + OFF_WHB);
    float* bias = (float*)(ws + OFF_BIAS);
    float* xg   = (float*)(ws + OFF_XG);
    bhalf* hbuf = (bhalf*)(ws + OFF_H);
    float* c    = (float*)(ws + OFF_C);

    // 1) precision conversion
    f32_to_bf16_kernel<<<16384, 256, 0, stream>>>(x,  xb,  B_ * S_ * DIN_);
    f32_to_bf16_kernel<<<4096,  256, 0, stream>>>(wx, wxb, HID_ * DIN_);
    f32_to_bf16_kernel<<<4096,  256, 0, stream>>>(wh, whb, HID_ * HS_);
    prep_kernel<<<(B_ * HS_ + 255) / 256, 256, 0, stream>>>(bx, bh, bias, hbuf, c);

    // 2) big input GEMM: 1024 M-tiles(32) x 64 N-tiles = 65536 waves / 4 per block
    xg_gemm_kernel<<<16384, 128, 0, stream>>>(xb, wxb, bias, xg);

    // 3) recurrence: stream ordering provides inter-step sync; h double-buffered
    for (int s = 0; s < S_; ++s) {
        const bhalf* h_in  = hbuf + (size_t)(s & 1) * B_ * HS_;
        bhalf*       h_out = hbuf + (size_t)((s + 1) & 1) * B_ * HS_;
        lstm_step_kernel<<<32, 256, 131072, stream>>>(xg, whb, h_in, h_out, c, out, s);
    }
}
